// DGCNN_18897856102402
// MI455X (gfx1250) — matrix operand order repaired and checked by reading, not compiled, not run on hardware
//
#include <hip/hip_runtime.h>
#include <hip/hip_bf16.h>

typedef __attribute__((ext_vector_type(16))) _Float16 v16h;
typedef __attribute__((ext_vector_type(8)))  float    v8f;

#define NPTS 1024
#define KNN  20

// ---------------------------------------------------------------------------
// squared norms xx[b,n] = sum_c x[b,c,n]^2
// ---------------------------------------------------------------------------
__global__ void sqnorm_kernel(const float* __restrict__ x, int totalC, int off,
                              int C, float* __restrict__ xx) {
  int b = blockIdx.y;
  int n = blockIdx.x * blockDim.x + threadIdx.x;
  if (n >= NPTS) return;
  const float* xb = x + ((size_t)b * totalC + off) * NPTS;
  float s = 0.0f;
  for (int c = 0; c < C; ++c) { float t = xb[c * NPTS + n]; s = fmaf(t, t, s); }
  xx[b * NPTS + n] = s;
}

// ---------------------------------------------------------------------------
// KNN: one block per (b, i). Full distance row in LDS, then 20 argmax passes.
// pd(i,j) = 2*dot(xi,xj) - |xi|^2 - |xj|^2  (self gets 0 == max, like JAX ref)
// ---------------------------------------------------------------------------
__global__ __launch_bounds__(256)
void knn_kernel(const float* __restrict__ x, int totalC, int off, int C,
                const float* __restrict__ xx, int* __restrict__ idxout) {
  __shared__ float dist[NPTS];
  __shared__ float ctr[128];
  __shared__ float rv[256];
  __shared__ int   ri[256];
  const int i = blockIdx.x, b = blockIdx.y, tid = threadIdx.x;
  const float* xb = x + ((size_t)b * totalC + off) * NPTS;
  if (tid < C) ctr[tid] = xb[tid * NPTS + i];
  __syncthreads();
  const float xxi = xx[b * NPTS + i];
  for (int j = tid; j < NPTS; j += 256) {
    float dot = 0.0f;
    for (int c = 0; c < C; ++c) dot = fmaf(ctr[c], xb[c * NPTS + j], dot);
    dist[j] = 2.0f * dot - xxi - xx[b * NPTS + j];
  }
  __syncthreads();
  for (int kk = 0; kk < KNN; ++kk) {
    float bv = -3.0e38f; int bi = NPTS;
    for (int j = tid; j < NPTS; j += 256) {
      float d = dist[j];
      if (d > bv || (d == bv && j < bi)) { bv = d; bi = j; }
    }
    rv[tid] = bv; ri[tid] = bi;
    __syncthreads();
    for (int sft = 128; sft > 0; sft >>= 1) {
      if (tid < sft) {
        float ov = rv[tid + sft]; int oi = ri[tid + sft];
        if (ov > rv[tid] || (ov == rv[tid] && oi < ri[tid])) { rv[tid] = ov; ri[tid] = oi; }
      }
      __syncthreads();
    }
    if (tid == 0) {
      idxout[((size_t)b * NPTS + i) * KNN + kk] = ri[0];
      dist[ri[0]] = -3.0e38f;
    }
    __syncthreads();
  }
}

// ---------------------------------------------------------------------------
// Fused edge conv: gather edge features + WMMA GEMM + BN + LeakyReLU + max_k.
// One wave per (n-tile of 16, co-tile of 16, batch). A = W tile (LDS),
// B = on-the-fly gathered edge features, D accumulated 20x (one per neighbor
// rank) with running max in registers. Never materializes (B,Cout,N,k).
// ---------------------------------------------------------------------------
template<int CIN, int COUT>
__global__ __launch_bounds__(32)
void edge_conv_wmma(const float* __restrict__ xin, int inTotalC, int inOff,
                    const int* __restrict__ knn,
                    const float* __restrict__ w,
                    const float* __restrict__ bs, const float* __restrict__ bo,
                    const float* __restrict__ bm, const float* __restrict__ bv,
                    float* __restrict__ outp, int outTotalC, int outOff) {
  constexpr int CK  = 2 * CIN;
  constexpr int NCH = (CK + 31) / 32;
  const int ntile = blockIdx.x;
  const int cot   = blockIdx.y;
  const int b     = blockIdx.z;
  const int lane  = threadIdx.x;
  const int nloc  = lane & 15;
  const int hi    = lane >> 4;

  __shared__ _Float16 sW[16 * NCH * 32];
  __shared__ float    sCtr[CIN * 16];
  __shared__ int      sIdx[16 * KNN];

  for (int t = lane; t < 16 * NCH * 32; t += 32) {
    int r = t / (NCH * 32);
    int c = t % (NCH * 32);
    sW[t] = (_Float16)((c < CK) ? w[(cot * 16 + r) * CK + c] : 0.0f);
  }
  const float* xb = xin + ((size_t)b * inTotalC + inOff) * NPTS;
  for (int t = lane; t < CIN * 16; t += 32)
    sCtr[t] = xb[(t >> 4) * NPTS + ntile * 16 + (t & 15)];
  for (int t = lane; t < 16 * KNN; t += 32)
    sIdx[t] = knn[((size_t)b * NPTS + ntile * 16 + t / KNN) * KNN + t % KNN];
  __syncthreads();

  float sc8[8], bb8[8];
  #pragma unroll
  for (int r = 0; r < 8; ++r) {
    int co = cot * 16 + r + hi * 8;
    float sc = bs[co] * rsqrtf(bv[co] + 1e-5f);
    sc8[r] = sc;
    bb8[r] = bo[co] - bm[co] * sc;
  }

  float mx[8];
  #pragma unroll
  for (int r = 0; r < 8; ++r) mx[r] = -3.0e38f;

  #pragma clang loop unroll(disable)
  for (int kk = 0; kk < KNN; ++kk) {
    const int j = sIdx[nloc * KNN + kk];
    const float* xj = xb + j;
    v8f acc = {};
    #pragma unroll
    for (int cc = 0; cc < NCH; ++cc) {
      v16h A, B;
      #pragma unroll
      for (int h = 0; h < 16; ++h) {
        // A-matrix 16-bit 16x32 layout: M = lane&15, K = h(+8 if h>=8)(+8 hi)
        int ka = (h < 8 ? h : h + 8) + hi * 8;
        A[h] = sW[nloc * (NCH * 32) + cc * 32 + ka];
        // B-matrix 32x16 layout: N = lane&15, K = h(+8 if h>=8)(+8 hi)
        int cg = cc * 32 + (h < 8 ? h : h + 8) + hi * 8;
        float val = 0.0f;
        if (cg < CIN)      val = xj[cg * NPTS] - sCtr[cg * 16 + nloc];
        else if (cg < CK)  val = sCtr[(cg - CIN) * 16 + nloc];
        B[h] = (_Float16)val;
      }
      acc = __builtin_amdgcn_wmma_f32_16x16x32_f16(false, A, false, B,
                                                   (short)0, acc, false, false);
    }
    #pragma unroll
    for (int r = 0; r < 8; ++r) {
      float y = acc[r] * sc8[r] + bb8[r];
      y = (y >= 0.0f) ? y : 0.2f * y;
      mx[r] = fmaxf(mx[r], y);
    }
  }
  const int n = ntile * 16 + nloc;
  #pragma unroll
  for (int r = 0; r < 8; ++r) {
    int co = cot * 16 + r + hi * 8;
    outp[((size_t)b * outTotalC + outOff + co) * NPTS + n] = mx[r];
  }
}

// ---------------------------------------------------------------------------
// Final 1x1 conv (1024 x 512) + BN + LeakyReLU + fused max/mean pool over N.
// One wave per (o-tile of 16, batch); loops all 64 n-tiles with WMMA and
// reduces max/sum across the 16 column lanes via shfl_xor (wave32 halves).
// ---------------------------------------------------------------------------
__global__ __launch_bounds__(32)
void conv5_pool(const float* __restrict__ cat, const float* __restrict__ w5,
                const float* __restrict__ bs, const float* __restrict__ bo,
                const float* __restrict__ bm, const float* __restrict__ bv,
                float* __restrict__ feat1, float* __restrict__ feat2) {
  constexpr int C = 512;
  const int ot   = blockIdx.x;
  const int b    = blockIdx.y;
  const int lane = threadIdx.x;
  const int nloc = lane & 15;
  const int hi   = lane >> 4;

  __shared__ _Float16 sW[16 * C];
  for (int t = lane; t < 16 * C; t += 32)
    sW[t] = (_Float16)w5[(size_t)(ot * 16 + t / C) * C + (t % C)];
  __syncthreads();

  float sc8[8], bb8[8];
  #pragma unroll
  for (int r = 0; r < 8; ++r) {
    int o = ot * 16 + r + hi * 8;
    float sc = bs[o] * rsqrtf(bv[o] + 1e-5f);
    sc8[r] = sc;
    bb8[r] = bo[o] - bm[o] * sc;
  }

  float mx[8], sm[8];
  #pragma unroll
  for (int r = 0; r < 8; ++r) { mx[r] = -3.0e38f; sm[r] = 0.0f; }

  const float* cb = cat + (size_t)b * C * NPTS;
  #pragma clang loop unroll(disable)
  for (int nt = 0; nt < NPTS / 16; ++nt) {
    v8f acc = {};
    #pragma unroll
    for (int cc = 0; cc < C / 32; ++cc) {
      v16h A, B;
      #pragma unroll
      for (int h = 0; h < 16; ++h) {
        int ka = (h < 8 ? h : h + 8) + hi * 8;
        A[h] = sW[nloc * C + cc * 32 + ka];
        int c = cc * 32 + (h < 8 ? h : h + 8) + hi * 8;
        B[h] = (_Float16)cb[(size_t)c * NPTS + nt * 16 + nloc];
      }
      acc = __builtin_amdgcn_wmma_f32_16x16x32_f16(false, A, false, B,
                                                   (short)0, acc, false, false);
    }
    #pragma unroll
    for (int r = 0; r < 8; ++r) {
      float y = acc[r] * sc8[r] + bb8[r];
      y = (y >= 0.0f) ? y : 0.2f * y;
      mx[r] = fmaxf(mx[r], y);
      sm[r] += y;
    }
  }
  #pragma unroll
  for (int r = 0; r < 8; ++r) {
    #pragma unroll
    for (int d = 1; d < 16; d <<= 1) {
      mx[r] = fmaxf(mx[r], __shfl_xor(mx[r], d, 32));
      sm[r] += __shfl_xor(sm[r], d, 32);
    }
  }
  if (nloc == 0) {
    #pragma unroll
    for (int r = 0; r < 8; ++r) {
      int o = ot * 16 + r + hi * 8;
      float p1 = mx[r], p2 = sm[r] * (1.0f / NPTS);
      feat1[b * 2048 + o] = p1;        feat1[b * 2048 + 1024 + o] = p2;
      feat2[b * 2048 + o] = p1;        feat2[b * 2048 + 1024 + o] = p2;
    }
  }
}

// ---------------------------------------------------------------------------
// FC1 + BN + ReLU : h = relu(bn(feat @ wl1.T + bl1)). M = batch = 16 exactly.
// ---------------------------------------------------------------------------
__global__ __launch_bounds__(32)
void mlp1_kernel(const float* __restrict__ feat, const float* __restrict__ wl1,
                 const float* __restrict__ bl1,
                 const float* __restrict__ s6, const float* __restrict__ o6,
                 const float* __restrict__ m6, const float* __restrict__ v6,
                 float* __restrict__ hr) {
  const int ot = blockIdx.x;
  const int lane = threadIdx.x, l15 = lane & 15, hi = lane >> 4;
  const int o = ot * 16 + l15;
  v8f acc = {};
  #pragma clang loop unroll(disable)
  for (int cc = 0; cc < 2048 / 32; ++cc) {
    v16h A, B;
    #pragma unroll
    for (int h = 0; h < 16; ++h) {
      int ka = (h < 8 ? h : h + 8) + hi * 8;
      A[h] = (_Float16)feat[l15 * 2048 + cc * 32 + ka];           // M = batch
      int kb = (h < 8 ? h : h + 8) + hi * 8;
      B[h] = (_Float16)wl1[(size_t)o * 2048 + cc * 32 + kb];      // N = o
    }
    acc = __builtin_amdgcn_wmma_f32_16x16x32_f16(false, A, false, B,
                                                 (short)0, acc, false, false);
  }
  const float sc = s6[o] * rsqrtf(v6[o] + 1e-5f);
  const float bb = o6[o] - m6[o] * sc;
  const float bias = bl1[o];
  #pragma unroll
  for (int r = 0; r < 8; ++r) {
    float y = (acc[r] + bias) * sc + bb;
    hr[(r + hi * 8) * 1024 + o] = fmaxf(y, 0.0f);
  }
}

// ---------------------------------------------------------------------------
// FC2 : inv = h @ wl2.T + bl2
// ---------------------------------------------------------------------------
__global__ __launch_bounds__(32)
void mlp2_kernel(const float* __restrict__ hr, const float* __restrict__ wl2,
                 const float* __restrict__ bl2, float* __restrict__ inv) {
  const int ot = blockIdx.x;
  const int lane = threadIdx.x, l15 = lane & 15, hi = lane >> 4;
  const int o = ot * 16 + l15;
  v8f acc = {};
  #pragma clang loop unroll(disable)
  for (int cc = 0; cc < 1024 / 32; ++cc) {
    v16h A, B;
    #pragma unroll
    for (int h = 0; h < 16; ++h) {
      int ka = (h < 8 ? h : h + 8) + hi * 8;
      A[h] = (_Float16)hr[l15 * 1024 + cc * 32 + ka];
      int kb = (h < 8 ? h : h + 8) + hi * 8;
      B[h] = (_Float16)wl2[(size_t)o * 1024 + cc * 32 + kb];
    }
    acc = __builtin_amdgcn_wmma_f32_16x16x32_f16(false, A, false, B,
                                                 (short)0, acc, false, false);
  }
  const float bias = bl2[o];
  #pragma unroll
  for (int r = 0; r < 8; ++r)
    inv[(r + hi * 8) * 256 + o] = acc[r] + bias;
}

// ---------------------------------------------------------------------------
extern "C" void kernel_launch(void* const* d_in, const int* in_sizes, int n_in,
                              void* d_out, int out_size, void* d_ws, size_t ws_size,
                              hipStream_t stream) {
  const float* x   = (const float*)d_in[0];
  const float* w1  = (const float*)d_in[1];
  const float* s1  = (const float*)d_in[2];
  const float* o1  = (const float*)d_in[3];
  const float* m1  = (const float*)d_in[4];
  const float* v1  = (const float*)d_in[5];
  const float* w2  = (const float*)d_in[6];
  const float* s2  = (const float*)d_in[7];
  const float* o2  = (const float*)d_in[8];
  const float* m2  = (const float*)d_in[9];
  const float* v2  = (const float*)d_in[10];
  const float* w3  = (const float*)d_in[11];
  const float* s3  = (const float*)d_in[12];
  const float* o3  = (const float*)d_in[13];
  const float* m3  = (const float*)d_in[14];
  const float* v3  = (const float*)d_in[15];
  const float* w4  = (const float*)d_in[16];
  const float* s4  = (const float*)d_in[17];
  const float* o4  = (const float*)d_in[18];
  const float* m4  = (const float*)d_in[19];
  const float* v4  = (const float*)d_in[20];
  const float* w5  = (const float*)d_in[21];
  const float* s5  = (const float*)d_in[22];
  const float* o5  = (const float*)d_in[23];
  const float* m5  = (const float*)d_in[24];
  const float* v5  = (const float*)d_in[25];
  const float* wl1 = (const float*)d_in[26];
  const float* bl1 = (const float*)d_in[27];
  const float* s6  = (const float*)d_in[28];
  const float* o6  = (const float*)d_in[29];
  const float* m6  = (const float*)d_in[30];
  const float* v6  = (const float*)d_in[31];
  const float* wl2 = (const float*)d_in[32];
  const float* bl2 = (const float*)d_in[33];

  // workspace layout (≈35 MB total)
  char* ws = (char*)d_ws;
  float* xx  = (float*)ws;                              // 16*1024 f32      (64 KB)
  int*   idx = (int*)(ws + (64u << 10));                // 16*1024*20 i32   (1.25 MB)
  float* cat = (float*)(ws + (64u << 10) + 1310720u);   // 16*512*1024 f32  (32 MB)
  float* hr  = (float*)(ws + (64u << 10) + 1310720u + 33554432u); // 16*1024 f32

  float* feat1 = (float*)d_out;                  // (16, 2048)
  float* inv   = (float*)d_out + 16 * 2048;      // (16, 256)
  float* feat2 = (float*)d_out + 16 * 2048 + 16 * 256; // (16, 2048)

  dim3 gSq(4, 16), gKnn(1024, 16);

  // layer 1: x (3ch) -> cat[0:64]
  sqnorm_kernel<<<gSq, 256, 0, stream>>>(x, 3, 0, 3, xx);
  knn_kernel<<<gKnn, 256, 0, stream>>>(x, 3, 0, 3, xx, idx);
  edge_conv_wmma<3, 64><<<dim3(64, 4, 16), 32, 0, stream>>>(
      x, 3, 0, idx, w1, s1, o1, m1, v1, cat, 512, 0);

  // layer 2: x1 (64ch @ cat+0) -> cat[64:128]
  sqnorm_kernel<<<gSq, 256, 0, stream>>>(cat, 512, 0, 64, xx);
  knn_kernel<<<gKnn, 256, 0, stream>>>(cat, 512, 0, 64, xx, idx);
  edge_conv_wmma<64, 64><<<dim3(64, 4, 16), 32, 0, stream>>>(
      cat, 512, 0, idx, w2, s2, o2, m2, v2, cat, 512, 64);

  // layer 3: x2 (64ch @ cat+64) -> cat[128:256]
  sqnorm_kernel<<<gSq, 256, 0, stream>>>(cat, 512, 64, 64, xx);
  knn_kernel<<<gKnn, 256, 0, stream>>>(cat, 512, 64, 64, xx, idx);
  edge_conv_wmma<64, 128><<<dim3(64, 8, 16), 32, 0, stream>>>(
      cat, 512, 64, idx, w3, s3, o3, m3, v3, cat, 512, 128);

  // layer 4: x3 (128ch @ cat+128) -> cat[256:512]
  sqnorm_kernel<<<gSq, 256, 0, stream>>>(cat, 512, 128, 128, xx);
  knn_kernel<<<gKnn, 256, 0, stream>>>(cat, 512, 128, 128, xx, idx);
  edge_conv_wmma<128, 256><<<dim3(64, 16, 16), 32, 0, stream>>>(
      cat, 512, 128, idx, w4, s4, o4, m4, v4, cat, 512, 256);

  // head
  conv5_pool<<<dim3(64, 16), 32, 0, stream>>>(cat, w5, s5, o5, m5, v5, feat1, feat2);
  mlp1_kernel<<<64, 32, 0, stream>>>(feat1, wl1, bl1, s6, o6, m6, v6, hr);
  mlp2_kernel<<<16, 32, 0, stream>>>(hr, wl2, bl2, inv);
}
